// SpatialEmbeddingCrossAttentionBlock_24335284699429
// MI455X (gfx1250) — matrix operand order repaired and checked by reading, not compiled, not run on hardware
//
#include <hip/hip_runtime.h>
#include <math.h>

#define B_     8
#define CIN    320
#define INNER  512
#define LX     1024
#define CTX    512
#define DTOK   64
#define HEADS  8
#define HD     64
#define FFD    2048
#define NGROUPS 32

typedef _Float16 f16;
typedef __attribute__((ext_vector_type(16))) _Float16 v16h;
typedef __attribute__((ext_vector_type(8)))  float    v8f;

union FragU { v16h h; uint4 u[2]; };

// A-matrix 16-bit fragment: lane row = lane%16; K runs [8*hi,+8) and [16+8*hi,+8)
__device__ __forceinline__ v16h ldsA(const f16* rowBase, int hi) {
  FragU f;
  f.u[0] = *(const uint4*)(rowBase + 8*hi);
  f.u[1] = *(const uint4*)(rowBase + 16 + 8*hi);
  return f.h;
}
// B-matrix 16-bit fragment: lane col = lane%16; K runs [8*hi,+8) and [16+8*hi,+8)
__device__ __forceinline__ v16h ldsB(const f16* colBase, int hi) {
  FragU f;
  f.u[0] = *(const uint4*)(colBase + 8*hi);
  f.u[1] = *(const uint4*)(colBase + 16 + 8*hi);
  return f.h;
}
#define WMMA(a,b,c) __builtin_amdgcn_wmma_f32_16x16x32_f16(false,(a),false,(b),(short)0,(c),false,false)

// ---- CDNA5 async global->LDS (16-byte chunk), ASYNCcnt tracked ------------
typedef int v4i_ __attribute__((vector_size(16)));
typedef __attribute__((address_space(1))) v4i_* g_v4p;
typedef __attribute__((address_space(3))) v4i_* l_v4p;

__device__ __forceinline__ void async_cp16(f16* lds_dst, const f16* gsrc) {
#if __has_builtin(__builtin_amdgcn_global_load_async_to_lds_b128)
  __builtin_amdgcn_global_load_async_to_lds_b128((g_v4p)gsrc, (l_v4p)lds_dst, 0, 0);
#else
  unsigned loff = (unsigned)(unsigned long long)(__attribute__((address_space(3))) f16*)lds_dst;
  asm volatile("global_load_async_to_lds_b128 %0, %1, off"
               :: "v"(loff), "v"(gsrc) : "memory");
#endif
}
__device__ __forceinline__ void wait_async0() {
#if __has_builtin(__builtin_amdgcn_s_wait_asynccnt)
  __builtin_amdgcn_s_wait_asynccnt(0);
#else
  asm volatile("s_wait_asynccnt 0x0" ::: "memory");
#endif
}

// ---------------------------------------------------------------------------
// f32 -> f16 bulk convert (n % 4 == 0)
// ---------------------------------------------------------------------------
__global__ __launch_bounds__(256) void cvt_k(const float* __restrict__ X,
                                             f16* __restrict__ Y, long n) {
  long i = ((long)blockIdx.x * 256 + threadIdx.x) * 4;
  if (i >= n) return;
  float4 v = *(const float4*)(X + i);
  union { f16 h[4]; uint2 u; } p;
  p.h[0] = (f16)v.x; p.h[1] = (f16)v.y; p.h[2] = (f16)v.z; p.h[3] = (f16)v.w;
  *(uint2*)(Y + i) = p.u;
}

// x (B, CIN, LX) f32  ->  xt (B, LX, CIN) f16
__global__ __launch_bounds__(256) void xpose_k(const float* __restrict__ X,
                                               f16* __restrict__ Y) {
  size_t idx = (size_t)blockIdx.x * 256 + threadIdx.x;   // b*CIN*LX
  int l = (int)(idx & (LX - 1));
  size_t t = idx >> 10;
  int cI = (int)(t % CIN);
  int b = (int)(t / CIN);
  Y[((size_t)b * LX + l) * CIN + cI] = (f16)X[idx];
}

// ---------------------------------------------------------------------------
// GEMM: for batch b, out(o,l) = sum_i W[o,i] * X[b,l,i]  (f16 in, f32 acc)
// 256 thr = 8 waves (2m x 4n); block tile 64(O) x 128(L); wave tile 32x32.
// Double-buffered async-to-LDS staging; rows clamped for L < 128 edge.
// mode 0: Y16 token-major (B,L,O) [+gelu]
// mode 1: Y32 token-major + optional f32 residual R (token-major)
// mode 2: Y32 channel-major (B,O,L) + residual R (channel-major)
// mode 3: Y16 channel-major (B,O,L)
// ---------------------------------------------------------------------------
__global__ __launch_bounds__(256) void gemm16_k(
    const f16* __restrict__ W, const f16* __restrict__ X,
    const float* __restrict__ bias, const float* __restrict__ R,
    f16* __restrict__ Y16, float* __restrict__ Y32,
    int O, int I, int L, int mode, int gelu)
{
  __shared__ __align__(16) f16 Wl[2][64][32];
  __shared__ __align__(16) f16 Xl[2][128][32];
  const int bz = blockIdx.z;
  const int o0 = blockIdx.y * 64;
  const int l0 = blockIdx.x * 128;
  const int tid = threadIdx.x;
  const int w = tid >> 5, lane = tid & 31;
  const int wm2 = w >> 2, wn4 = w & 3;       // 2 x 4 wave grid
  const int ln16 = lane & 15, hi = lane >> 4;
  const f16* Xb = X + (size_t)bz * L * I;

  const int r = tid >> 2, part = tid & 3;    // chunks of 8 halfs
  auto stage = [&](int bu, int k0) {
    async_cp16(&Wl[bu][r][part * 8], W + (size_t)(o0 + r) * I + k0 + part * 8);
    int x0 = l0 + r;       if (x0 >= L) x0 = L - 1;
    int x1 = l0 + r + 64;  if (x1 >= L) x1 = L - 1;
    async_cp16(&Xl[bu][r][part * 8],      Xb + (size_t)x0 * I + k0 + part * 8);
    async_cp16(&Xl[bu][r + 64][part * 8], Xb + (size_t)x1 * I + k0 + part * 8);
  };

  v8f acc[2][2] = {};
  const int nk = I >> 5;
  stage(0, 0);
  for (int it = 0; it < nk; ++it) {
    wait_async0();
    __syncthreads();
    if (it + 1 < nk) stage((it + 1) & 1, (it + 1) << 5);
    const int cur = it & 1;
    v16h a0 = ldsA(&Wl[cur][wm2*32 + ln16][0], hi);
    v16h a1 = ldsA(&Wl[cur][wm2*32 + 16 + ln16][0], hi);
#pragma unroll
    for (int nt = 0; nt < 2; ++nt) {
      v16h bb = ldsB(&Xl[cur][wn4*32 + nt*16 + ln16][0], hi);
      acc[0][nt] = WMMA(a0, bb, acc[0][nt]);
      acc[1][nt] = WMMA(a1, bb, acc[1][nt]);
    }
  }

#pragma unroll
  for (int mt = 0; mt < 2; ++mt) {
#pragma unroll
    for (int nt = 0; nt < 2; ++nt) {
      const int l = l0 + wn4*32 + nt*16 + ln16;
      if (l >= L) continue;
      const int ob = o0 + wm2*32 + mt*16 + 8*hi;   // 8 consecutive out-channels
      float tmp[8];
#pragma unroll
      for (int rr = 0; rr < 8; ++rr) {
        float v = acc[mt][nt][rr] + bias[ob + rr];
        if (gelu) v = 0.5f * v * (1.0f + erff(v * 0.70710678118654752f));
        tmp[rr] = v;
      }
      if (mode == 0) {
        union { f16 h[8]; uint4 u; } p;
#pragma unroll
        for (int rr = 0; rr < 8; ++rr) p.h[rr] = (f16)tmp[rr];
        *(uint4*)(Y16 + ((size_t)bz * L + l) * O + ob) = p.u;
      } else if (mode == 1) {
        float* dst = Y32 + ((size_t)bz * L + l) * O + ob;
        if (R) {
          const float4* rp = (const float4*)(R + ((size_t)bz * L + l) * O + ob);
          float4 ra = rp[0], rb = rp[1];
          tmp[0] += ra.x; tmp[1] += ra.y; tmp[2] += ra.z; tmp[3] += ra.w;
          tmp[4] += rb.x; tmp[5] += rb.y; tmp[6] += rb.z; tmp[7] += rb.w;
        }
        float4 ya = {tmp[0], tmp[1], tmp[2], tmp[3]};
        float4 yb = {tmp[4], tmp[5], tmp[6], tmp[7]};
        *(float4*)dst = ya; *(float4*)(dst + 4) = yb;
      } else if (mode == 2) {
#pragma unroll
        for (int rr = 0; rr < 8; ++rr) {
          size_t a = ((size_t)bz * O + ob + rr) * L + l;
          Y32[a] = tmp[rr] + R[a];
        }
      } else {
#pragma unroll
        for (int rr = 0; rr < 8; ++rr)
          Y16[((size_t)bz * O + ob + rr) * L + l] = (f16)tmp[rr];
      }
    }
  }
}

// ---------------------------------------------------------------------------
// c_proj: bandwidth-bound (Wci is 64MB read once) -> warp-per-row VALU dot.
// Output token-major f32: cc[b, tok, ch],  o = ch*DTOK + tok.
// ---------------------------------------------------------------------------
__global__ __launch_bounds__(256) void cproj_k(
    const float* __restrict__ Wci, const float* __restrict__ c,
    const float* __restrict__ bci, float* __restrict__ cc)
{
  const int o = (blockIdx.x * 256 + threadIdx.x) >> 5;
  const int lane = threadIdx.x & 31;
  float acc[B_];
#pragma unroll
  for (int b = 0; b < B_; ++b) acc[b] = 0.f;
  for (int i = lane; i < CTX; i += 32) {
    float wv = Wci[(size_t)o * CTX + i];
#pragma unroll
    for (int b = 0; b < B_; ++b) acc[b] += wv * c[b * CTX + i];
  }
#pragma unroll
  for (int b = 0; b < B_; ++b) {
    float a = acc[b];
    a += __shfl_down(a, 16, 32);
    a += __shfl_down(a, 8, 32);
    a += __shfl_down(a, 4, 32);
    a += __shfl_down(a, 2, 32);
    a += __shfl_down(a, 1, 32);
    acc[b] = a;
  }
  if (lane == 0) {
    const int ch = o >> 6, tok = o & 63;
#pragma unroll
    for (int b = 0; b < B_; ++b)
      cc[((size_t)b * DTOK + tok) * CTX + ch] = acc[b] + bci[o];
  }
}

// ---------------------------------------------------------------------------
// GroupNorm, token-major (B,L,C) f32 in -> f16 out; group = 16 channels.
// ---------------------------------------------------------------------------
__global__ __launch_bounds__(256) void gnt_k(
    const float* __restrict__ X, f16* __restrict__ Y,
    const float* __restrict__ gw, const float* __restrict__ gb, int L, int C)
{
  const int g = blockIdx.x, b = blockIdx.y;
  const int chg = 16;                       // C / NGROUPS
  const int n = L * chg;
  const float* Xb = X + (size_t)b * L * C + g * chg;
  f16* Yb = Y + (size_t)b * L * C + g * chg;
  float s = 0.f, s2 = 0.f;
  for (int i = threadIdx.x; i < n; i += 256) {
    int l = i >> 4, cI = i & 15;
    float v = Xb[(size_t)l * C + cI];
    s += v; s2 += v * v;
  }
#pragma unroll
  for (int off = 16; off > 0; off >>= 1) {
    s  += __shfl_down(s,  off, 32);
    s2 += __shfl_down(s2, off, 32);
  }
  __shared__ float rs[8], rs2[8];
  __shared__ float mu, rsig;
  const int wv = threadIdx.x >> 5, lane = threadIdx.x & 31;
  if (lane == 0) { rs[wv] = s; rs2[wv] = s2; }
  __syncthreads();
  if (threadIdx.x == 0) {
    float a = 0.f, a2 = 0.f;
    for (int i = 0; i < 8; ++i) { a += rs[i]; a2 += rs2[i]; }
    float m = a / n;
    mu = m;
    rsig = rsqrtf(fmaxf(a2 / n - m * m, 0.f) + 1e-5f);
  }
  __syncthreads();
  for (int i = threadIdx.x; i < n; i += 256) {
    int l = i >> 4, cI = i & 15;
    float v = Xb[(size_t)l * C + cI];
    Yb[(size_t)l * C + cI] = (f16)((v - mu) * rsig * gw[g*chg + cI] + gb[g*chg + cI]);
  }
}

// ---------------------------------------------------------------------------
// Fused flash attention. Q,K token-major f16 (B,L,INNER); V channel-major f16
// (B,INNER,Lk). Out token-major f16. 2 waves/block, each owns a 16-query tile.
// All LDS tiles staged with async-to-LDS; K/V double-buffered; per-wave LDS.
// ---------------------------------------------------------------------------
__global__ __launch_bounds__(64) void attn2_k(
    const f16* __restrict__ Q, const f16* __restrict__ K,
    const f16* __restrict__ Vc, f16* __restrict__ Aout, int Lq, int Lk)
{
  __shared__ __align__(16) f16 Ql[2][16][64];
  __shared__ __align__(16) f16 Kl[2][2][32][64];
  __shared__ __align__(16) f16 Vl[2][2][64][32];
  __shared__ __align__(16) f16 Pl[2][16][32];
  const int w = threadIdx.x >> 5, lane = threadIdx.x & 31;
  const int ln16 = lane & 15, hi = lane >> 4;
  const int qb = blockIdx.x * 2 + w;
  const int hd = blockIdx.y, b = blockIdx.z;
  const f16* qrow = Q + ((size_t)b * Lq + qb * 16) * INNER + hd * HD;

#pragma unroll
  for (int t = 0; t < 4; ++t) {                 // Q tile: 16x64 = 128 chunks
    int idx = lane + t * 32;
    int pos = idx >> 3, part = idx & 7;
    async_cp16(&Ql[w][pos][part * 8], qrow + (size_t)pos * INNER + part * 8);
  }
  auto stageKV = [&](int bu, int kb) {
#pragma unroll
    for (int t = 0; t < 8; ++t) {
      int idx = lane + t * 32;
      int key = idx >> 3, part = idx & 7;       // K tile 32x64
      async_cp16(&Kl[w][bu][key][part * 8],
                 K + ((size_t)b * Lk + kb + key) * INNER + hd * HD + part * 8);
      int ch = idx >> 2, p2 = idx & 3;          // V tile 64x32 (channel-major src)
      async_cp16(&Vl[w][bu][ch][p2 * 8],
                 Vc + ((size_t)b * INNER + hd * HD + ch) * Lk + kb + p2 * 8);
    }
  };
  stageKV(0, 0);
  wait_async0();

  v16h qa0 = ldsA(&Ql[w][ln16][0], hi);
  v16h qa1 = ldsA(&Ql[w][ln16][32], hi);

  v8f oacc[4] = {};
  float mrow[8], lrow[8];
#pragma unroll
  for (int rr = 0; rr < 8; ++rr) { mrow[rr] = -1e30f; lrow[rr] = 0.f; }

  const int nkb = Lk >> 5;
  for (int it = 0; it < nkb; ++it) {
    const int cur = it & 1;
    if (it + 1 < nkb) stageKV(cur ^ 1, (it + 1) << 5);

    v8f s0 = {}, s1 = {};
    {
      v16h b0 = ldsB(&Kl[w][cur][ln16][0],  hi);       s0 = WMMA(qa0, b0, s0);
      v16h b1 = ldsB(&Kl[w][cur][ln16][32], hi);       s0 = WMMA(qa1, b1, s0);
      v16h b2 = ldsB(&Kl[w][cur][16 + ln16][0],  hi);  s1 = WMMA(qa0, b2, s1);
      v16h b3 = ldsB(&Kl[w][cur][16 + ln16][32], hi);  s1 = WMMA(qa1, b3, s1);
    }
#pragma unroll
    for (int rr = 0; rr < 8; ++rr) {
      float a0 = s0[rr] * 0.125f, a1 = s1[rr] * 0.125f;
      float mx = fmaxf(a0, a1);
      mx = fmaxf(mx, __shfl_xor(mx, 8, 32));
      mx = fmaxf(mx, __shfl_xor(mx, 4, 32));
      mx = fmaxf(mx, __shfl_xor(mx, 2, 32));
      mx = fmaxf(mx, __shfl_xor(mx, 1, 32));
      float mn = fmaxf(mrow[rr], mx);
      float al = expf(mrow[rr] - mn);
      float p0 = expf(a0 - mn), p1 = expf(a1 - mn);
      float ps = p0 + p1;
      ps += __shfl_xor(ps, 8, 32);
      ps += __shfl_xor(ps, 4, 32);
      ps += __shfl_xor(ps, 2, 32);
      ps += __shfl_xor(ps, 1, 32);
      lrow[rr] = lrow[rr] * al + ps;
      mrow[rr] = mn;
      Pl[w][rr + 8*hi][ln16]      = (f16)p0;
      Pl[w][rr + 8*hi][16 + ln16] = (f16)p1;
#pragma unroll
      for (int nt = 0; nt < 4; ++nt) oacc[nt][rr] *= al;
    }
    v16h pa = ldsA(&Pl[w][ln16][0], hi);
#pragma unroll
    for (int nt = 0; nt < 4; ++nt) {
      v16h vb = ldsB(&Vl[w][cur][nt*16 + ln16][0], hi);
      oacc[nt] = WMMA(pa, vb, oacc[nt]);
    }
    if (it + 1 < nkb) wait_async0();
  }
#pragma unroll
  for (int nt = 0; nt < 4; ++nt) {
#pragma unroll
    for (int rr = 0; rr < 8; ++rr) {
      size_t addr = ((size_t)b * Lq + qb * 16 + rr + 8*hi) * INNER
                  + hd * HD + nt*16 + ln16;
      Aout[addr] = (f16)(oacc[nt][rr] / lrow[rr]);
    }
  }
}

// ---------------------------------------------------------------------------
extern "C" void kernel_launch(void* const* d_in, const int* in_sizes, int n_in,
                              void* d_out, int out_size, void* d_ws, size_t ws_size,
                              hipStream_t stream)
{
  (void)in_sizes; (void)n_in; (void)out_size; (void)ws_size;
  const float* x   = (const float*)d_in[0];
  const float* c   = (const float*)d_in[1];
  const float* Wxi = (const float*)d_in[2];  const float* bxi = (const float*)d_in[3];
  const float* Wci = (const float*)d_in[4];  const float* bci = (const float*)d_in[5];
  const float* Wq1 = (const float*)d_in[6];  const float* bq1 = (const float*)d_in[7];
  const float* Wk1 = (const float*)d_in[8];  const float* bk1 = (const float*)d_in[9];
  const float* Wv1 = (const float*)d_in[10]; const float* bv1 = (const float*)d_in[11];
  const float* Wo1 = (const float*)d_in[12]; const float* bo1 = (const float*)d_in[13];
  const float* Wq2 = (const float*)d_in[14]; const float* bq2 = (const float*)d_in[15];
  const float* Wk2 = (const float*)d_in[16]; const float* bk2 = (const float*)d_in[17];
  const float* Wv2 = (const float*)d_in[18]; const float* bv2 = (const float*)d_in[19];
  const float* Wo2 = (const float*)d_in[20]; const float* bo2 = (const float*)d_in[21];
  const float* gnxw = (const float*)d_in[22]; const float* gnxb = (const float*)d_in[23];
  const float* gncw = (const float*)d_in[24]; const float* gncb = (const float*)d_in[25];
  const float* Wf1 = (const float*)d_in[26]; const float* bf1 = (const float*)d_in[27];
  const float* Wf2 = (const float*)d_in[28]; const float* bf2 = (const float*)d_in[29];
  const float* Wxo = (const float*)d_in[30]; const float* bxo = (const float*)d_in[31];
  float* out = (float*)d_out;

  char* wsb = (char*)d_ws;
  const size_t MB = 1048576ull;
  float* h    = (float*)(wsb);                 // (B,L,512) f32   16MB
  f16*  hn    = (f16*)  (wsb + 16*MB);         // (B,L,512) f16    8MB
  f16*  q16   = (f16*)  (wsb + 24*MB);         // (B,L,512) f16    8MB
  f16*  k16   = (f16*)  (wsb + 32*MB);         // (B,L,512) f16    8MB
  f16*  v16   = (f16*)  (wsb + 40*MB);         // (B,512,Lk) f16   8MB
  f16*  ab16  = (f16*)  (wsb + 48*MB);         // (B,L,512) f16    8MB
  f16*  t1    = (f16*)  (wsb + 24*MB);         // (B,L,2048) f16  32MB (reuse q..ab)
  float* ccr  = (float*)(wsb + 56*MB);         // (B,64,512) f32   1MB
  f16*  cc16  = (f16*)  (wsb + 57*MB);         // (B,64,512) f16  .5MB
  f16*  xt16  = (f16*)  (wsb + 58*MB);         // (B,L,320) f16   5.25MB
  f16*  wb    = (f16*)  (wsb + 64*MB);         // f16 weights     ~8.7MB

  f16* Wxi16 = wb;                 size_t off = (size_t)INNER * CIN;
  f16* Wq116 = wb + off;           off += (size_t)INNER * INNER;
  f16* Wk116 = wb + off;           off += (size_t)INNER * INNER;
  f16* Wv116 = wb + off;           off += (size_t)INNER * INNER;
  f16* Wo116 = wb + off;           off += (size_t)INNER * INNER;
  f16* Wq216 = wb + off;           off += (size_t)INNER * INNER;
  f16* Wk216 = wb + off;           off += (size_t)INNER * CTX;
  f16* Wv216 = wb + off;           off += (size_t)INNER * CTX;
  f16* Wo216 = wb + off;           off += (size_t)INNER * INNER;
  f16* Wf116 = wb + off;           off += (size_t)FFD * INNER;
  f16* Wf216 = wb + off;           off += (size_t)INNER * FFD;
  f16* Wxo16 = wb + off;

  dim3 blk(256);
  auto cvt = [&](const float* src, f16* dst, long n) {
    cvt_k<<<dim3((unsigned)(n / 4 / 256)), blk, 0, stream>>>(src, dst, n);
  };
  // weights -> f16 (once per launch)
  cvt(Wxi, Wxi16, (long)INNER * CIN);
  cvt(Wq1, Wq116, (long)INNER * INNER);
  cvt(Wk1, Wk116, (long)INNER * INNER);
  cvt(Wv1, Wv116, (long)INNER * INNER);
  cvt(Wo1, Wo116, (long)INNER * INNER);
  cvt(Wq2, Wq216, (long)INNER * INNER);
  cvt(Wk2, Wk216, (long)INNER * CTX);
  cvt(Wv2, Wv216, (long)INNER * CTX);
  cvt(Wo2, Wo216, (long)INNER * INNER);
  cvt(Wf1, Wf116, (long)FFD * INNER);
  cvt(Wf2, Wf216, (long)INNER * FFD);
  cvt(Wxo, Wxo16, (long)CIN * INNER);

  xpose_k<<<dim3(B_ * CIN * LX / 256), blk, 0, stream>>>(x, xt16);

  dim3 gHL(LX / 128, INNER / 64, B_);
  dim3 gKV(1, INNER / 64, B_);                 // L = DTOK = 64 (row-clamped)
  // x_proj_in: h = Wxi @ x + bxi   (f32 token-major)
  gemm16_k<<<gHL, blk, 0, stream>>>(Wxi16, xt16, bxi, nullptr, nullptr, h,
                                    INNER, CIN, LX, 1, 0);
  // context branch
  cproj_k<<<dim3((CTX * DTOK) / 8), blk, 0, stream>>>(Wci, c, bci, ccr);
  gnt_k<<<dim3(NGROUPS, B_), blk, 0, stream>>>(ccr, cc16, gncw, gncb, DTOK, CTX);

  // ---- self attention ----
  gnt_k<<<dim3(NGROUPS, B_), blk, 0, stream>>>(h, hn, gnxw, gnxb, LX, INNER);
  gemm16_k<<<gHL, blk, 0, stream>>>(Wq116, hn, bq1, nullptr, q16, nullptr, INNER, INNER, LX, 0, 0);
  gemm16_k<<<gHL, blk, 0, stream>>>(Wk116, hn, bk1, nullptr, k16, nullptr, INNER, INNER, LX, 0, 0);
  gemm16_k<<<gHL, blk, 0, stream>>>(Wv116, hn, bv1, nullptr, v16, nullptr, INNER, INNER, LX, 3, 0);
  attn2_k<<<dim3(LX / 32, HEADS, B_), dim3(64), 0, stream>>>(q16, k16, v16, ab16, LX, LX);
  gemm16_k<<<gHL, blk, 0, stream>>>(Wo116, ab16, bo1, h, nullptr, h, INNER, INNER, LX, 1, 0);

  // ---- cross attention ----
  gnt_k<<<dim3(NGROUPS, B_), blk, 0, stream>>>(h, hn, gnxw, gnxb, LX, INNER);
  gemm16_k<<<gHL, blk, 0, stream>>>(Wq216, hn, bq2, nullptr, q16, nullptr, INNER, INNER, LX, 0, 0);
  gemm16_k<<<gKV, blk, 0, stream>>>(Wk216, cc16, bk2, nullptr, k16, nullptr, INNER, CTX, DTOK, 0, 0);
  gemm16_k<<<gKV, blk, 0, stream>>>(Wv216, cc16, bv2, nullptr, v16, nullptr, INNER, CTX, DTOK, 3, 0);
  attn2_k<<<dim3(LX / 32, HEADS, B_), dim3(64), 0, stream>>>(q16, k16, v16, ab16, LX, DTOK);
  gemm16_k<<<gHL, blk, 0, stream>>>(Wo216, ab16, bo2, h, nullptr, h, INNER, INNER, LX, 1, 0);

  // ---- feed forward (exact GELU) ----
  gnt_k<<<dim3(NGROUPS, B_), blk, 0, stream>>>(h, hn, gnxw, gnxb, LX, INNER);
  gemm16_k<<<dim3(LX / 128, FFD / 64, B_), blk, 0, stream>>>(Wf116, hn, bf1, nullptr, t1, nullptr, FFD, INNER, LX, 0, 1);
  gemm16_k<<<gHL, blk, 0, stream>>>(Wf216, t1, bf2, h, nullptr, h, INNER, FFD, LX, 1, 0);

  // ---- x_proj_out + outer residual (channel-major f32 output) ----
  cvt(h, hn, (long)B_ * LX * INNER);
  gemm16_k<<<dim3(LX / 128, CIN / 64, B_), blk, 0, stream>>>(Wxo16, hn, bxo, x, nullptr, out, CIN, INNER, LX, 2, 0);
}
